// CrystalGNN_45500883533883
// MI455X (gfx1250) — matrix operand order repaired and checked by reading, not compiled, not run on hardware
//
#include <hip/hip_runtime.h>
#include <hip/hip_bf16.h>

// ---------------------------------------------------------------------------
// CGConv GNN on MI455X (gfx1250):
//   edge MLPs via v_wmma_f32_16x16x32_bf16, edge_attr staged by TDM
//   (tensor_load_to_lds with LDS padding), x gathers as b128 copies of a
//   pre-converted bf16 node table.
// ---------------------------------------------------------------------------

typedef __attribute__((ext_vector_type(16))) __bf16   v16bf;
typedef __attribute__((ext_vector_type(8)))  float    v8f;
typedef __attribute__((ext_vector_type(4)))  unsigned u32x4;
typedef __attribute__((ext_vector_type(4)))  unsigned v4u;
typedef __attribute__((ext_vector_type(8)))  int      v8i;
typedef __attribute__((ext_vector_type(4)))  int      v4i;

#define N_NODES   20000
#define N_EDGES   320000
#define NODE_DIM  64
#define HIDDEN    128
#define N_GRAPHS  64
#define N_TILES   (N_EDGES / 16)          // 20000 edge tiles of 16
#define WAVES_PER_BLOCK 4
#define TILES_PER_WAVE  8
#define EDGE_BLOCKS (N_TILES / (WAVES_PER_BLOCK * TILES_PER_WAVE))  // 625

#if __has_builtin(__builtin_amdgcn_tensor_load_to_lds) && \
    __has_builtin(__builtin_amdgcn_s_wait_tensorcnt)
#define USE_TDM 1
#endif

// round-to-nearest-even f32 -> bf16 (bits)
__device__ __forceinline__ unsigned f2bf(float f) {
  unsigned u = __float_as_uint(f);
  return (u + 0x7FFFu + ((u >> 16) & 1u)) >> 16;
}
__device__ __forceinline__ unsigned packbf(float lo, float hi) {
  return f2bf(lo) | (f2bf(hi) << 16);
}

// ---------------------------------------------------------------------------
// f32 -> packed bf16 pair stream conversion
// ---------------------------------------------------------------------------
__global__ void cvt_bf16_kernel(const float* __restrict__ src,
                                unsigned* __restrict__ dst, int npairs) {
  int i = blockIdx.x * blockDim.x + threadIdx.x;
  if (i < npairs) dst[i] = packbf(src[2 * i], src[2 * i + 1]);
}

// ---------------------------------------------------------------------------
// Pack the 4 weight matrices (256x64 f32, row-major W[k][n]) into BF16
// B-fragment order for V_WMMA_F32_16X16X32_BF16:
//   idx = ((w*4 + nTile)*8 + kStep)*32*8 + lane*8 + v   (dwords)
//   lane 0-15 : col n = nTile*16+lane,      K = kStep*32 + v*2 + {0,1}
//   lane 16-31: col n = nTile*16+(lane-16), K = kStep*32 + 16 + v*2 + {0,1}
// ---------------------------------------------------------------------------
__global__ void pack_weights_kernel(const float* __restrict__ Wf1,
                                    const float* __restrict__ Ws1,
                                    const float* __restrict__ Wf2,
                                    const float* __restrict__ Ws2,
                                    unsigned* __restrict__ out) {
  int idx = blockIdx.x * blockDim.x + threadIdx.x;
  if (idx >= 4 * 8192) return;
  int w     = idx >> 13;
  int r     = idx & 8191;
  int v     = r & 7;
  int lane  = (r >> 3) & 31;
  int kStep = (r >> 8) & 7;
  int nTile = (r >> 11) & 3;
  const float* W = (w == 0) ? Wf1 : (w == 1) ? Ws1 : (w == 2) ? Wf2 : Ws2;
  int n = nTile * 16 + (lane & 15);
  int k = kStep * 32 + ((lane >> 4) << 4) + v * 2;
  out[idx] = packbf(W[k * NODE_DIM + n], W[(k + 1) * NODE_DIM + n]);
}

__global__ void copy_f32_kernel(const float* __restrict__ src,
                                float* __restrict__ dst, int n) {
  int i = blockIdx.x * blockDim.x + threadIdx.x;
  if (i < n) dst[i] = src[i];
}

__global__ void zero_f32_kernel(float* __restrict__ p, int n) {
  int i = blockIdx.x * blockDim.x + threadIdx.x;
  if (i < n) p[i] = 0.0f;
}

// ---------------------------------------------------------------------------
// Edge kernel: one wave per 16-edge tile.
// LDS z row layout (512 B/row): [x_dst 128B | x_src 128B | edge_attr 256B]
//   - x halves gathered as b128 copies from bf16 node table
//   - edge_attr block loaded by ONE tensor_load_to_lds per tile with LDS
//     padding (write 64 dwords, skip 64 dwords) to interleave into the rows
//   gate = sigmoid(z@Wf + bf), soft = softplus(z@Ws + bs)
//   atomic scatter-add gate*soft into acc[dst]  (acc pre-initialized to x)
// ---------------------------------------------------------------------------
__global__ void __launch_bounds__(WAVES_PER_BLOCK * 32)
cgconv_edge_kernel(const unsigned* __restrict__ xbf,   // nodes, 32 dwords/row
                   const unsigned* __restrict__ eabf,  // edges, 64 dwords/row
                   const int*      __restrict__ ei,    // [2, E]
                   const unsigned* __restrict__ wPack, // 16384 dwords
                   const float*    __restrict__ bgate,
                   const float*    __restrict__ bsoft,
                   float*          __restrict__ acc) {
  extern __shared__ char smem[];
  unsigned* ldsW   = (unsigned*)smem;                    // 16384 dwords
  unsigned* ldsZ   = (unsigned*)(smem + 65536);          // 4 * 2048 dwords
  int*      ldsIdx = (int*)(smem + 65536 + 32768);       // 4 * 32 ints

  for (int i = threadIdx.x; i < 16384; i += blockDim.x) ldsW[i] = wPack[i];
  __syncthreads();

  const int lane = threadIdx.x & 31;
  const int wave = threadIdx.x >> 5;
  unsigned* zbuf  = ldsZ + wave * 2048;
  u32x4*    zq    = (u32x4*)zbuf;                        // 16B chunks
  int*      idxbf = ldsIdx + wave * 32;
  const u32x4* xq = (const u32x4*)xbf;                   // 8 chunks / node row

  // hoist per-lane biases (column = nTile*16 + lane&15)
  float bgv[4], bsv[4];
#pragma unroll
  for (int nt = 0; nt < 4; ++nt) {
    int col = nt * 16 + (lane & 15);
    bgv[nt] = bgate[col];
    bsv[nt] = bsoft[col];
  }

  const int row   = lane & 15;
  const int khalf = lane >> 4;
  const int tile0 = (blockIdx.x * WAVES_PER_BLOCK + wave) * TILES_PER_WAVE;

#ifdef USE_TDM
  // wave-invariant parts of the tensor descriptor
  const unsigned zoff = (unsigned)(size_t)(void*)zbuf;   // LDS byte offset
#endif

  for (int t = 0; t < TILES_PER_WAVE; ++t) {
    const int tile = tile0 + t;
    if (tile >= N_TILES) break;
    const int ebase = tile * 16;

#ifdef USE_TDM
    {
      // TDM: 2D tile 128 elem x 16 rows of bf16 from eabf, landing at
      // zoff+256 with pad: 64 dwords written (row's ea), 64 dwords skipped.
      unsigned long long ga =
          (unsigned long long)(size_t)eabf + (unsigned long long)ebase * 256ull;
      v4u g0;
      g0.x = 1u;                                           // count=1
      g0.y = zoff + 256u;                                  // lds_addr
      g0.z = (unsigned)(ga & 0xFFFFFFFFull);               // global_addr lo
      g0.w = (unsigned)((ga >> 32) & 0x01FFFFFFull) | 0x80000000u; // hi|type=2
      v8i g1;
      g1[0] = (int)((1u << 16) |                           // data_size = 2B
                    (1u << 20) |                           // pad_enable
                    (5u << 22) |                           // pad_interval: 64 dw
                    (63u << 25));                          // pad_amount: 64 dw
      g1[1] = (int)((128u & 0xFFFFu) << 16);               // tensor_dim0 lo16
      g1[2] = (int)(((unsigned)N_EDGES & 0xFFFFu) << 16);  // dim0 hi=0|dim1 lo
      g1[3] = (int)(((unsigned)N_EDGES >> 16) |            // tensor_dim1 hi16
                    (128u << 16));                         // tile_dim0 = 128
      g1[4] = 16;                                          // tile_dim1=16
      g1[5] = 128;                                         // dim0_stride lo32
      g1[6] = 0;
      g1[7] = 0;
      v4i gz4 = {0, 0, 0, 0};
      v8i gz8 = {0, 0, 0, 0, 0, 0, 0, 0};
      // 6-arg toolchain form: (g0, g1, g2, g3, g4, cpol)
      __builtin_amdgcn_tensor_load_to_lds(g0, g1, gz4, gz4, gz8, 0);
    }
#endif

    // stage the 16 src/dst indices
    if (lane < 16) {
      idxbf[lane]      = ei[ebase + lane];            // src
      idxbf[16 + lane] = ei[N_EDGES + ebase + lane];  // dst
    }

    // gather x_dst / x_src halves: 256 chunks of 16B, 8 per lane
#pragma unroll
    for (int j = 0; j < 8; ++j) {
      int c  = lane + j * 32;
      int er = c >> 4, q = c & 15;
      long node = (q < 8) ? idxbf[16 + er] : idxbf[er];
      zq[er * 32 + q] = xq[node * 8 + (q & 7)];
    }

#ifndef USE_TDM
    // fallback: copy edge_attr block (16 rows x 16 chunks of 16B)
    {
      const u32x4* eaq = (const u32x4*)eabf;
#pragma unroll
      for (int j = 0; j < 8; ++j) {
        int c  = lane + j * 32;
        int er = c >> 4, q = c & 15;
        zq[er * 32 + 16 + q] = eaq[(long)(ebase + er) * 16 + q];
      }
    }
#endif

#ifdef USE_TDM
    __builtin_amdgcn_s_wait_tensorcnt((short)0);
#endif
    // same-wave DS RAW is in-order; compiler inserts s_wait_dscnt for loads

    v8f accG[4] = {v8f{}, v8f{}, v8f{}, v8f{}};
    v8f accS[4] = {v8f{}, v8f{}, v8f{}, v8f{}};

#pragma unroll
    for (int ks = 0; ks < 8; ++ks) {
      // A fragment (16x32 bf16): lanes 0-15 K=0..15, lanes 16-31 K=16..31
      union { u32x4 q[2]; v16bf v; } A;
      const int ab = (row * 128 + ks * 16 + khalf * 8) >> 2;  // 16B units
      A.q[0] = zq[ab];
      A.q[1] = zq[ab + 1];
#pragma unroll
      for (int nt = 0; nt < 4; ++nt) {
        union { u32x4 q[2]; v16bf v; } BG, BS;
        const u32x4* pg = (const u32x4*)(ldsW +
                          (((nt * 8 + ks) * 32 + lane) << 3));
        const u32x4* ps = (const u32x4*)(ldsW + 8192 +
                          (((nt * 8 + ks) * 32 + lane) << 3));
        BG.q[0] = pg[0]; BG.q[1] = pg[1];
        BS.q[0] = ps[0]; BS.q[1] = ps[1];
        accG[nt] = __builtin_amdgcn_wmma_f32_16x16x32_bf16(
            false, A.v, false, BG.v, (short)0, accG[nt], false, false);
        accS[nt] = __builtin_amdgcn_wmma_f32_16x16x32_bf16(
            false, A.v, false, BS.v, (short)0, accS[nt], false, false);
      }
    }

    // epilogue: bias + sigmoid*softplus, atomic scatter-add to acc[dst]
#pragma unroll
    for (int nt = 0; nt < 4; ++nt) {
      const int col = nt * 16 + (lane & 15);
#pragma unroll
      for (int r = 0; r < 8; ++r) {
        const int m = r + khalf * 8;           // edge row per C/D layout
        const int d = idxbf[16 + m];
        float g = accG[nt][r] + bgv[nt];
        float s = accS[nt][r] + bsv[nt];
        float sig = 1.0f / (1.0f + __expf(-g));
        float sp  = (s > 20.0f) ? s : __logf(1.0f + __expf(s));
        __hip_atomic_fetch_add(acc + (long)d * NODE_DIM + col, sig * sp,
                               __ATOMIC_RELAXED, __HIP_MEMORY_SCOPE_AGENT);
      }
    }
  }
}

// ---------------------------------------------------------------------------
// Pooling: segment-sum nodes into graphs
// ---------------------------------------------------------------------------
__global__ void pool_sum_kernel(const float* __restrict__ x,
                                const int* __restrict__ batch,
                                float* __restrict__ sums,
                                float* __restrict__ cnt) {
  int idx = blockIdx.x * blockDim.x + threadIdx.x;
  if (idx >= N_NODES * NODE_DIM) return;
  int node = idx >> 6, c = idx & 63;
  int g = batch[node];
  __hip_atomic_fetch_add(sums + g * NODE_DIM + c, x[idx],
                         __ATOMIC_RELAXED, __HIP_MEMORY_SCOPE_AGENT);
  if (c == 0)
    __hip_atomic_fetch_add(cnt + g, 1.0f,
                           __ATOMIC_RELAXED, __HIP_MEMORY_SCOPE_AGENT);
}

// ---------------------------------------------------------------------------
// Head: pooled = sums/cnt; h = relu(pooled@W1+b1); out = h@W2+b2  (tiny)
// ---------------------------------------------------------------------------
__global__ void __launch_bounds__(128)
head_kernel(const float* __restrict__ sums, const float* __restrict__ cnt,
            const float* __restrict__ W1, const float* __restrict__ b1,
            const float* __restrict__ W2, const float* __restrict__ b2,
            float* __restrict__ out) {
  __shared__ float pooled[N_GRAPHS][NODE_DIM];
  __shared__ float h[N_GRAPHS][HIDDEN];
  const int tid = threadIdx.x;
  for (int i = tid; i < N_GRAPHS * NODE_DIM; i += 128) {
    int g = i >> 6, c = i & 63;
    float cc = cnt[g];
    cc = (cc < 1.0f) ? 1.0f : cc;
    pooled[g][c] = sums[i] / cc;
  }
  __syncthreads();
  for (int g = 0; g < N_GRAPHS; ++g) {
    float a = b1[tid];
#pragma unroll 8
    for (int c = 0; c < NODE_DIM; ++c) a += pooled[g][c] * W1[c * HIDDEN + tid];
    h[g][tid] = (a > 0.0f) ? a : 0.0f;
  }
  __syncthreads();
  for (int g = tid; g < N_GRAPHS; g += 128) {
    float a = b2[0];
#pragma unroll 8
    for (int k = 0; k < HIDDEN; ++k) a += h[g][k] * W2[k];
    out[g] = a;
  }
}

// ---------------------------------------------------------------------------
extern "C" void kernel_launch(void* const* d_in, const int* in_sizes, int n_in,
                              void* d_out, int out_size, void* d_ws, size_t ws_size,
                              hipStream_t stream) {
  const float* x     = (const float*)d_in[0];
  const int*   ei    = (const int*)d_in[1];
  const float* ea    = (const float*)d_in[2];
  const int*   batch = (const int*)d_in[3];
  const float* Wf1 = (const float*)d_in[4];
  const float* bf1 = (const float*)d_in[5];
  const float* Ws1 = (const float*)d_in[6];
  const float* bs1 = (const float*)d_in[7];
  const float* Wf2 = (const float*)d_in[8];
  const float* bf2 = (const float*)d_in[9];
  const float* Ws2 = (const float*)d_in[10];
  const float* bs2 = (const float*)d_in[11];
  const float* W1  = (const float*)d_in[12];
  const float* b1  = (const float*)d_in[13];
  const float* W2  = (const float*)d_in[14];
  const float* b2  = (const float*)d_in[15];
  float* out = (float*)d_out;

  char* ws = (char*)d_ws;
  unsigned* wPack = (unsigned*)ws;                          //       131,072 B
  float* accA = (float*)(ws + 131072);                      //     5,120,000 B
  float* accB = (float*)(ws + 5251072);                     //     5,120,000 B
  float* sums = (float*)(ws + 10371072);                    //        16,384 B
  float* cnt  = sums + N_GRAPHS * NODE_DIM;                 //           256 B
  unsigned* xbf  = (unsigned*)(ws + 10387712);              //     2,560,000 B
  unsigned* eabf = (unsigned*)(ws + 12947712);              //    81,920,000 B

  const int nxf   = N_NODES * NODE_DIM;                     // 1,280,000
  const int xprs  = nxf / 2;                                // bf16 pairs in x
  const int eaprs = N_EDGES * HIDDEN / 2;                   // 20,480,000
  const size_t smem = 65536 + 32768 + 512;                  // 98,816 B LDS

  // one-time conversions (re-run every call: ws is not preserved)
  pack_weights_kernel<<<(4 * 8192 + 255) / 256, 256, 0, stream>>>(
      Wf1, Ws1, Wf2, Ws2, wPack);
  cvt_bf16_kernel<<<(eaprs + 255) / 256, 256, 0, stream>>>(ea, eabf, eaprs);

  // layer 1: acc = x + scatter(m)
  cvt_bf16_kernel<<<(xprs + 255) / 256, 256, 0, stream>>>(x, xbf, xprs);
  copy_f32_kernel<<<(nxf + 255) / 256, 256, 0, stream>>>(x, accA, nxf);
  cgconv_edge_kernel<<<EDGE_BLOCKS, WAVES_PER_BLOCK * 32, smem, stream>>>(
      xbf, eabf, ei, wPack, bf1, bs1, accA);

  // layer 2: acc = x1 + scatter(m)
  cvt_bf16_kernel<<<(xprs + 255) / 256, 256, 0, stream>>>(accA, xbf, xprs);
  copy_f32_kernel<<<(nxf + 255) / 256, 256, 0, stream>>>(accA, accB, nxf);
  cgconv_edge_kernel<<<EDGE_BLOCKS, WAVES_PER_BLOCK * 32, smem, stream>>>(
      xbf, eabf, ei, wPack + 16384, bf2, bs2, accB);

  // global mean pool + fc head
  zero_f32_kernel<<<(N_GRAPHS * NODE_DIM + N_GRAPHS + 255) / 256, 256, 0, stream>>>(
      sums, N_GRAPHS * NODE_DIM + N_GRAPHS);
  pool_sum_kernel<<<(nxf + 255) / 256, 256, 0, stream>>>(accB, batch, sums, cnt);
  head_kernel<<<1, 128, 0, stream>>>(sums, cnt, W1, b1, W2, b2, out);
}
